// MultiPartGCNModel_19104014532917
// MI455X (gfx1250) — compile-verified, matrix operand order checked by reading, not run on hardware
//
#include <hip/hip_runtime.h>
#include <hip/hip_bf16.h>
#include <stdint.h>

// ---------- types ----------
typedef unsigned short u16;
typedef __bf16 bf16_t;
typedef u16    v8u  __attribute__((ext_vector_type(8)));
typedef u16    v16u __attribute__((ext_vector_type(16)));
typedef bf16_t v16bf __attribute__((ext_vector_type(16)));
typedef float  v8f  __attribute__((ext_vector_type(8)));

#define NPART 5
#define B_    32
#define T_    256
#define KTOT  274
#define PROJ  64
#define EMBED 256
#define KLMAX 137
#define TPAD  (T_ + 4)   // 260, SAME padding for KT=5 temporal conv

__device__ __forceinline__ u16 f2bf(float f) {
    unsigned int x = __builtin_bit_cast(unsigned int, f);
    unsigned int r = (x + 0x7FFFu + ((x >> 16) & 1u)) >> 16;
    return (u16)r;
}

// Per-lane K-slice loader for 16-bit WMMA operands (A and B share the pattern):
// lanes 0-15 hold K {0..7, 16..23}; lanes 16-31 hold K {8..15, 24..31}.
// p = this lane's base pointer at K=0 of its 32-K window (K contiguous in memory).
__device__ __forceinline__ v16bf load_kslice(const u16* __restrict__ p, int half) {
    v8u lo = *reinterpret_cast<const v8u*>(p + half * 8);
    v8u hi = *reinterpret_cast<const v8u*>(p + 16 + half * 8);
    v16u t;
#pragma unroll
    for (int i = 0; i < 8; ++i) { t[i] = lo[i]; t[i + 8] = hi[i]; }
    return __builtin_bit_cast(v16bf, t);
}

// ---------- weight repack: W_t (P,64,64,5) -> wA2[p][dout][tau*64+din] bf16 ----------
__global__ void repack_wt(const float* __restrict__ W_t, u16* __restrict__ wA2) {
    int tid = blockIdx.x * blockDim.x + threadIdx.x;
    int n = NPART * PROJ * PROJ * 5;
    if (tid >= n) return;
    int k   = tid % (PROJ * 5);          // tau*64+din
    int m   = (tid / (PROJ * 5)) % PROJ; // dout
    int p   = tid / (PROJ * 5 * PROJ);
    int tau = k >> 6;
    int din = k & 63;
    wA2[tid] = f2bf(W_t[(((size_t)p * PROJ + m) * PROJ + din) * 5 + tau]);
}

// ---------- weight repack: W_out (P,256,64) -> bf16, same layout ----------
__global__ void repack_wo(const float* __restrict__ W_out, u16* __restrict__ wA3) {
    int tid = blockIdx.x * blockDim.x + threadIdx.x;
    int n = NPART * EMBED * PROJ;
    if (tid >= n) return;
    wA3[tid] = f2bf(W_out[tid]);
}

// ---------- zero the temporal pad rows of h1 [b][j][{0,1,258,259}][64] ----------
__global__ void zeropad(u16* __restrict__ h1) {
    int tid = blockIdx.x * blockDim.x + threadIdx.x;
    int n = B_ * KLMAX * 4 * 8;
    if (tid >= n) return;
    int d8 = tid & 7; int rest = tid >> 3;
    int tp = rest & 3; rest >>= 2;
    int j = rest % KLMAX; int b = rest / KLMAX;
    int t = (tp < 2) ? tp : (256 + tp);  // 0,1,258,259
    v8u z = {};
    *reinterpret_cast<v8u*>(h1 + (((size_t)b * KLMAX + j) * TPAD + t) * PROJ + d8 * 8) = z;
}

// ---------- Aeff = adj + ares ----------
__global__ void addmat(const float* __restrict__ a, const float* __restrict__ r,
                       float* __restrict__ dst, int n) {
    int tid = blockIdx.x * blockDim.x + threadIdx.x;
    if (tid < n) dst[tid] = a[tid] + r[tid];
}

// ---------- sA[j] = sum_k Aeff[k][j] ----------
__global__ void colsum(const float* __restrict__ Aeff, float* __restrict__ sA,
                       int kl, int aoff, int soff) {
    int j = blockIdx.x * blockDim.x + threadIdx.x;
    if (j >= kl) return;
    float s = 0.f;
    for (int k = 0; k < kl; ++k) s += Aeff[aoff + (size_t)k * kl + j];
    sA[soff + j] = s;
}

// ---------- stage 1a: xa[b,t,j,c] = sum_k x[b,t,k,c] * Aeff[k,j] ----------
__global__ void stage1a(const float* __restrict__ pose, const float* __restrict__ Aeff,
                        float* __restrict__ xa, int kl, int joff, int aoff) {
    int tid = blockIdx.x * blockDim.x + threadIdx.x;
    int n = B_ * T_ * kl;
    if (tid >= n) return;
    int j  = tid % kl;
    int bt = tid / kl;
    const float* xp = pose + ((size_t)bt * KTOT + joff) * 3;
    const float* ac = Aeff + aoff + j;
    float a0 = 0.f, a1 = 0.f;
    for (int k = 0; k < kl; ++k) {
        float w = ac[(size_t)k * kl];
        a0 += xp[k * 3 + 0] * w;
        a1 += xp[k * 3 + 1] * w;
    }
    xa[(size_t)tid * 2 + 0] = a0;
    xa[(size_t)tid * 2 + 1] = a1;
}

// ---------- stage 1b: h1[b][j][t+2][d] = bf16(relu(w0*xa0 + w1*xa1 + bi*sA)) ----------
__global__ void stage1b(const float* __restrict__ xa, const float* __restrict__ sA,
                        const float* __restrict__ W_in, const float* __restrict__ b_in,
                        u16* __restrict__ h1, int part, int kl, int soff) {
    int tid = blockIdx.x * blockDim.x + threadIdx.x;
    int n = B_ * T_ * kl * 8;
    if (tid >= n) return;
    int d8 = tid & 7; int rest = tid >> 3;
    int j = rest % kl; int bt = rest / kl;
    int b = bt >> 8;   int t = bt & 255;
    float x0 = xa[(size_t)rest * 2 + 0];
    float x1 = xa[(size_t)rest * 2 + 1];
    float s  = sA[soff + j];
    v8u pk;
#pragma unroll
    for (int i = 0; i < 8; ++i) {
        int d = d8 * 8 + i;
        float w0 = W_in[((size_t)part * PROJ + d) * 2 + 0];
        float w1 = W_in[((size_t)part * PROJ + d) * 2 + 1];
        float f = w0 * x0 + w1 * x1 + b_in[part * PROJ + d] * s;
        pk[i] = f2bf(f > 0.f ? f : 0.f);
    }
    *reinterpret_cast<v8u*>(h1 + (((size_t)b * KLMAX + j) * TPAD + (t + 2)) * PROJ + d8 * 8) = pk;
}

// ---------- stage 2: temporal conv as WMMA GEMM (M=64, K=320, N=t) + relu + mean_j ----------
// One wave handles a 16-t strip; loops j, 10 K-tiles, 4 M-tiles => 40 WMMAs / j.
__global__ __launch_bounds__(256) void stage2_conv(
    const u16* __restrict__ h1, const u16* __restrict__ wA2,
    const float* __restrict__ b_t, u16* __restrict__ hbar, int part, int kl) {
    const int lane = threadIdx.x & 31;
    const int wid  = threadIdx.x >> 5;
    const int half = lane >> 4;
    const int col  = lane & 15;
    const int b    = blockIdx.y;
    const int t0   = blockIdx.x * 128 + wid * 16;

    float btv[4][8];
#pragma unroll
    for (int mt = 0; mt < 4; ++mt)
#pragma unroll
        for (int v = 0; v < 8; ++v)
            btv[mt][v] = b_t[part * PROJ + mt * 16 + 8 * half + v];

    float macc[4][8];
#pragma unroll
    for (int mt = 0; mt < 4; ++mt)
#pragma unroll
        for (int v = 0; v < 8; ++v) macc[mt][v] = 0.f;

    for (int j = 0; j < kl; ++j) {
        v8f c[4];
#pragma unroll
        for (int mt = 0; mt < 4; ++mt)
#pragma unroll
            for (int v = 0; v < 8; ++v) c[mt][v] = 0.f;

        const u16* hrow = h1 + (((size_t)b * KLMAX + j) * TPAD + (t0 + col)) * PROJ;
#pragma unroll
        for (int kt = 0; kt < 10; ++kt) {
            const int k0   = kt * 32;
            const int tau  = k0 >> 6;
            const int din0 = k0 & 63;
            v16bf bm = load_kslice(hrow + tau * PROJ + din0, half);
#pragma unroll
            for (int mt = 0; mt < 4; ++mt) {
                const u16* ap = wA2 + ((size_t)part * PROJ + mt * 16 + col) * 320 + k0;
                v16bf am = load_kslice(ap, half);
                c[mt] = __builtin_amdgcn_wmma_f32_16x16x32_bf16(
                    false, am, false, bm, (short)0, c[mt], false, false);
            }
        }
#pragma unroll
        for (int mt = 0; mt < 4; ++mt)
#pragma unroll
            for (int v = 0; v < 8; ++v) {
                float r = c[mt][v] + btv[mt][v];
                macc[mt][v] += (r > 0.f ? r : 0.f);
            }
    }

    const float inv = 1.0f / (float)kl;
    u16* outp = hbar + (((size_t)part * B_ + b) * T_ + (t0 + col)) * PROJ;
#pragma unroll
    for (int mt = 0; mt < 4; ++mt) {
        v8u pk;
#pragma unroll
        for (int v = 0; v < 8; ++v) pk[v] = f2bf(macc[mt][v] * inv);
        *reinterpret_cast<v8u*>(outp + mt * 16 + 8 * half) = pk;
    }
}

// ---------- stage 3: out[b,p,t,e] = hbar·W_out^T + b_out, masked (WMMA M=256,K=64) ----------
__global__ __launch_bounds__(128) void stage3_out(
    const u16* __restrict__ hbar, const u16* __restrict__ wA3,
    const float* __restrict__ b_out, const unsigned char* __restrict__ mask,
    float* __restrict__ out) {
    const int lane = threadIdx.x & 31;
    const int wid  = threadIdx.x >> 5;   // e-group: 4 waves * 64 = 256
    const int half = lane >> 4;
    const int col  = lane & 15;
    const int t0   = blockIdx.x * 16;
    const int b    = blockIdx.y;
    const int part = blockIdx.z;

    v8f c[4];
#pragma unroll
    for (int mt = 0; mt < 4; ++mt)
#pragma unroll
        for (int v = 0; v < 8; ++v) c[mt][v] = 0.f;

    const u16* hrow = hbar + (((size_t)part * B_ + b) * T_ + (t0 + col)) * PROJ;
#pragma unroll
    for (int kt = 0; kt < 2; ++kt) {
        const int k0 = kt * 32;
        v16bf bm = load_kslice(hrow + k0, half);
#pragma unroll
        for (int mt = 0; mt < 4; ++mt) {
            const int e_row = wid * 64 + mt * 16 + col;
            const u16* ap = wA3 + ((size_t)part * EMBED + e_row) * PROJ + k0;
            v16bf am = load_kslice(ap, half);
            c[mt] = __builtin_amdgcn_wmma_f32_16x16x32_bf16(
                false, am, false, bm, (short)0, c[mt], false, false);
        }
    }

    const int t = t0 + col;
    const float mk = mask[b * T_ + t] ? 1.0f : 0.0f;
#pragma unroll
    for (int mt = 0; mt < 4; ++mt)
#pragma unroll
        for (int v = 0; v < 8; ++v) {
            int e = wid * 64 + mt * 16 + 8 * half + v;
            out[(((size_t)b * NPART + part) * T_ + t) * EMBED + e] =
                (c[mt][v] + b_out[part * EMBED + e]) * mk;
        }
}

// ---------- launch ----------
extern "C" void kernel_launch(void* const* d_in, const int* in_sizes, int n_in,
                              void* d_out, int out_size, void* d_ws, size_t ws_size,
                              hipStream_t stream) {
    (void)in_sizes; (void)n_in; (void)out_size; (void)ws_size;
    const float* pose = (const float*)d_in[0];
    const unsigned char* mask = (const unsigned char*)d_in[1];
    const float* adj[NPART]  = {(const float*)d_in[2], (const float*)d_in[3], (const float*)d_in[4],
                                (const float*)d_in[5], (const float*)d_in[6]};
    const float* ares[NPART] = {(const float*)d_in[7], (const float*)d_in[8], (const float*)d_in[9],
                                (const float*)d_in[10], (const float*)d_in[11]};
    const float* W_in  = (const float*)d_in[12];
    const float* b_in  = (const float*)d_in[13];
    const float* W_t   = (const float*)d_in[14];
    const float* b_t   = (const float*)d_in[15];
    const float* W_out = (const float*)d_in[16];
    const float* b_out = (const float*)d_in[17];
    float* out = (float*)d_out;

    static const int kls[NPART]  = {25, 70, 21, 21, 137};
    static const int joff[NPART] = {0, 25, 95, 116, 137};
    static const int aoff[NPART] = {0, 625, 5525, 5966, 6407};  // prefix sums of kl^2
    const int AEFF_TOT = 25176;

    // workspace partition (256B aligned regions)
    char* base = (char*)d_ws;
    size_t o = 0;
    auto take = [&](size_t bytes) { char* p = base + o; o += (bytes + 255) & ~(size_t)255; return p; };
    float* Aeff = (float*)take((size_t)AEFF_TOT * 4);
    float* sA   = (float*)take((size_t)KTOT * 4);
    u16*   wA2  = (u16*)take((size_t)NPART * PROJ * 320 * 2);
    u16*   wA3  = (u16*)take((size_t)NPART * EMBED * PROJ * 2);
    float* xa   = (float*)take((size_t)B_ * T_ * KLMAX * 2 * 4);
    u16*   h1   = (u16*)take((size_t)B_ * KLMAX * TPAD * PROJ * 2);  // reused per part
    u16*   hbar = (u16*)take((size_t)NPART * B_ * T_ * PROJ * 2);

    // weight repack + pad zeroing
    {
        int n = NPART * PROJ * PROJ * 5;
        repack_wt<<<(n + 255) / 256, 256, 0, stream>>>(W_t, wA2);
    }
    {
        int n = NPART * EMBED * PROJ;
        repack_wo<<<(n + 255) / 256, 256, 0, stream>>>(W_out, wA3);
    }
    {
        int n = B_ * KLMAX * 4 * 8;
        zeropad<<<(n + 255) / 256, 256, 0, stream>>>(h1);
    }

    for (int p = 0; p < NPART; ++p) {
        const int kl = kls[p];
        {
            int n = kl * kl;
            addmat<<<(n + 255) / 256, 256, 0, stream>>>(adj[p], ares[p], Aeff + aoff[p], n);
        }
        colsum<<<(kl + 63) / 64, 64, 0, stream>>>(Aeff, sA, kl, aoff[p], joff[p]);
        {
            int n = B_ * T_ * kl;
            stage1a<<<(n + 255) / 256, 256, 0, stream>>>(pose, Aeff, xa, kl, joff[p], aoff[p]);
        }
        {
            int n = B_ * T_ * kl * 8;
            stage1b<<<(n + 255) / 256, 256, 0, stream>>>(xa, sA, W_in, b_in, h1, p, kl, joff[p]);
        }
        stage2_conv<<<dim3(T_ / 128, B_), 256, 0, stream>>>(h1, wA2, b_t, hbar, p, kl);
    }

    stage3_out<<<dim3(T_ / 16, B_, NPART), 128, 0, stream>>>(hbar, wA3, b_out, mask, out);
}
